// LinearTransformerModel_48069273977589
// MI455X (gfx1250) — matrix operand order repaired and checked by reading, not compiled, not run on hardware
//
#include <hip/hip_runtime.h>
#include <hip/hip_bf16.h>
#include <math.h>

typedef __attribute__((ext_vector_type(16))) __bf16 v16bf;
typedef __attribute__((ext_vector_type(8)))  float  v8f;
typedef __attribute__((ext_vector_type(4)))  int    v4i;

#define DEV __device__ __forceinline__

#if __has_builtin(__builtin_amdgcn_global_load_async_to_lds_b128)
#define HAVE_ASYNC 1
#else
#define HAVE_ASYNC 0
#endif

namespace {

constexpr int Bn    = 16;
constexpr int Tn    = 1024;
constexpr int Dn    = 512;
constexpr int Ln    = 6;
constexpr int FFn   = 2048;
constexpr int Mrows = Bn * Tn;   // 16384 (multiple of 128)

DEV unsigned short f2bf(float f) {
  unsigned u = __float_as_uint(f);
  u += 0x7FFFu + ((u >> 16) & 1u);          // round-to-nearest-even
  return (unsigned short)(u >> 16);
}
DEV float bf2f(unsigned short h) {
  return __uint_as_float(((unsigned)h) << 16);
}

#if HAVE_ASYNC
typedef __attribute__((address_space(1))) v4i gv4i;   // global int4
typedef __attribute__((address_space(3))) v4i lv4i;   // LDS int4
DEV void async_copy16(const unsigned short* g, unsigned short* l) {
  // GLOBAL_LOAD_ASYNC_TO_LDS_B128: per-lane 16B global -> LDS, ASYNCcnt-tracked
  __builtin_amdgcn_global_load_async_to_lds_b128((gv4i*)g, (lv4i*)l, 0, 0);
}
DEV void wait_async_0() {
#if __has_builtin(__builtin_amdgcn_s_wait_asynccnt)
  __builtin_amdgcn_s_wait_asynccnt(0);
#else
  asm volatile("s_wait_asynccnt 0x0" ::: "memory");
#endif
}
DEV void wait_async_4() {
#if __has_builtin(__builtin_amdgcn_s_wait_asynccnt)
  __builtin_amdgcn_s_wait_asynccnt(4);
#else
  asm volatile("s_wait_asynccnt 0x4" ::: "memory");
#endif
}
#endif

union FragAB { uint4 q[2]; v16bf v; };

// ---------------------------------------------------------------------------
// Weight prep: fp32 W[K x N] -> bf16 Wt[Npad x K] (transposed, zero-padded)
// ---------------------------------------------------------------------------
__global__ __launch_bounds__(256)
void convert_transpose(const float* __restrict__ W, unsigned short* __restrict__ Wt,
                       int K, int N, int Npad) {
  const int idx = blockIdx.x * 256 + threadIdx.x;
  if (idx >= Npad * K) return;
  const int n = idx % Npad;           // coalesced read along n
  const int k = idx / Npad;
  Wt[(size_t)n * K + k] = (n < N) ? f2bf(W[(size_t)k * N + n]) : (unsigned short)0;
}

// ---------------------------------------------------------------------------
// GEMM: C[M x N] = A[M x K](bf16) @ Bt[N x K](bf16, pre-transposed) (+bias)(+gelu)
// Double-buffered LDS tiles filled with CDNA5 async global->LDS b128 copies.
// 256 threads = 8 waves (4 x 2), wave tile 32x64, 8 WMMA accumulators.
// ---------------------------------------------------------------------------
template<bool BIAS, bool GELU, bool OUTBF>
__global__ __launch_bounds__(256)
void gemm_bf16_wmma(const unsigned short* __restrict__ A,
                    const unsigned short* __restrict__ Bt,
                    const float* __restrict__ bias,
                    float* __restrict__ Cf, unsigned short* __restrict__ Cb,
                    int N, int K) {
  __shared__ __align__(16) unsigned short As[2][128][32];   // [M][K]
  __shared__ __align__(16) unsigned short Bs[2][128][32];   // [N][K]

  const int tid  = threadIdx.x;
  const int lane = tid & 31;
  const int wave = tid >> 5;
  const int wr   = wave >> 1;      // 0..3 : wave row (32 rows)
  const int wcl  = wave & 1;       // 0..1 : wave col (64 cols)
  const int r    = lane & 15;
  const int hi   = lane >> 4;
  const int m0   = blockIdx.y * 128;
  const int n0   = blockIdx.x * 128;

  v8f acc[2][4];
  #pragma unroll
  for (int i = 0; i < 2; ++i)
    #pragma unroll
    for (int j = 0; j < 4; ++j)
      #pragma unroll
      for (int e = 0; e < 8; ++e) acc[i][j][e] = 0.f;

  // Each 128x32 bf16 tile = 512 x 16B chunks; 2 chunks per thread per matrix.
  auto stage = [&](int buf, int kk) {
    #pragma unroll
    for (int hh = 0; hh < 2; ++hh) {
      const int c   = tid + hh * 256;
      const int row = c >> 2;
      const int seg = (c & 3) * 8;
      const unsigned short* ga = A  + (size_t)(m0 + row) * K + kk + seg;
      const unsigned short* gb = Bt + (size_t)(n0 + row) * K + kk + seg;
#if HAVE_ASYNC
      async_copy16(ga, &As[buf][row][seg]);
      async_copy16(gb, &Bs[buf][row][seg]);
#else
      *(uint4*)&As[buf][row][seg] = *(const uint4*)ga;
      *(uint4*)&Bs[buf][row][seg] = *(const uint4*)gb;
#endif
    }
  };

  stage(0, 0);
  int buf = 0;
  for (int kk = 0; kk < K; kk += 32, buf ^= 1) {
    const bool hasNext = (kk + 32) < K;
    if (hasNext) stage(buf ^ 1, kk + 32);   // prefetch next tile (other buffer)
#if HAVE_ASYNC
    if (hasNext) wait_async_4();            // in-order: current tile landed
    else         wait_async_0();
#endif
    __syncthreads();

    FragAB af[2], bfq[4];
    #pragma unroll
    for (int i = 0; i < 2; ++i) {
      const unsigned short* ap = &As[buf][wr * 32 + i * 16 + r][0];
      af[i].q[0] = *(const uint4*)(ap + hi * 8);        // K = hi*8 .. +7
      af[i].q[1] = *(const uint4*)(ap + hi * 8 + 16);   // K = 16+hi*8 .. +7
    }
    #pragma unroll
    for (int j = 0; j < 4; ++j) {
      const unsigned short* bp = &Bs[buf][wcl * 64 + j * 16 + r][0];
      bfq[j].q[0] = *(const uint4*)(bp + hi * 8);       // K = hi*8 .. +7
      bfq[j].q[1] = *(const uint4*)(bp + hi * 8 + 16);  // K = 16+hi*8 .. +7
    }
    #pragma unroll
    for (int i = 0; i < 2; ++i)
      #pragma unroll
      for (int j = 0; j < 4; ++j)
        acc[i][j] = __builtin_amdgcn_wmma_f32_16x16x32_bf16(
            false, af[i].v, false, bfq[j].v, (short)0, acc[i][j], false, false);
    __syncthreads();
  }

  // epilogue: C/D layout -> lane(0-15): M=e, lane(16-31): M=8+e
  #pragma unroll
  for (int j = 0; j < 4; ++j) {
    const int col = n0 + wcl * 64 + j * 16 + r;
    if (col >= N) continue;
    const float bv = BIAS ? bias[col] : 0.f;
    #pragma unroll
    for (int i = 0; i < 2; ++i) {
      #pragma unroll
      for (int e = 0; e < 8; ++e) {
        const int row = m0 + wr * 32 + i * 16 + hi * 8 + e;
        float v = acc[i][j][e] + bv;
        if (GELU) v = 0.5f * v * (1.f + erff(v * 0.70710678118f));
        if (OUTBF) Cb[(size_t)row * N + col] = f2bf(v);
        else       Cf[(size_t)row * N + col] = v;
      }
    }
  }
}

// ---------------------------------------------------------------------------
// Embedding gather -> bf16 xin[row][0:224], keep mask
// ---------------------------------------------------------------------------
__global__ __launch_bounds__(224)
void embed_kernel(const int* __restrict__ mcc, const int* __restrict__ ab,
                  const int* __restrict__ hr,
                  const float* __restrict__ e_mcc, const float* __restrict__ e_ab,
                  const float* __restrict__ e_hr,
                  unsigned short* __restrict__ xin, int* __restrict__ keep) {
  const int row = blockIdx.x;
  const int tid = threadIdx.x;
  const int im = mcc[row];
  if (tid == 0) keep[row] = (im != 0) ? 1 : 0;
  float v;
  if (tid < 128)       v = e_mcc[im * 128 + tid];
  else if (tid < 192)  v = e_ab[ab[row] * 64 + (tid - 128)];
  else                 v = e_hr[hr[row] * 32 + (tid - 192)];
  xin[(size_t)row * 224 + tid] = f2bf(v);
}

// ---------------------------------------------------------------------------
// Positional encoding added in place; also emits bf16 copy for next GEMM A
// ---------------------------------------------------------------------------
__global__ __launch_bounds__(256)
void add_pos(float* __restrict__ x, unsigned short* __restrict__ xb) {
  const int idx = blockIdx.x * 256 + threadIdx.x;   // over Mrows*Dn
  const int col = idx & (Dn - 1);
  const int row = idx >> 9;
  const int t   = row & (Tn - 1);
  const int i2  = col & ~1;
  const float ang = (float)t * __expf(-9.21034037198f * (float)i2 / (float)Dn);
  const float v = x[idx] + ((col & 1) ? __cosf(ang) : __sinf(ang));
  x[idx]  = v;
  xb[idx] = f2bf(v);
}

// ---------------------------------------------------------------------------
// Local windowed attention, heads 0..3; writes bf16 output.
// ---------------------------------------------------------------------------
__global__ __launch_bounds__(128)
void local_attn(const float* __restrict__ qb, const float* __restrict__ kb,
                const float* __restrict__ vb, const int* __restrict__ keep,
                unsigned short* __restrict__ attn) {
  __shared__ unsigned short kch[128][64];
  __shared__ unsigned short vch[128][64];
  __shared__ int kmask[128];  // 0 = pad, 1 = valid&!keep, 3 = valid&keep

  const int u = blockIdx.x, h = blockIdx.y, b = blockIdx.z;
  const int i = threadIdx.x;
  const int t = u * 128 + i;
  const size_t qoff = ((size_t)(b * Tn + t)) * 512 + h * 64;

  float q[64], acc[64];
  #pragma unroll
  for (int e = 0; e < 64; ++e) { q[e] = qb[qoff + e]; acc[e] = 0.f; }
  const int keepq = keep[b * Tn + t];
  float m = -3.0e38f, s = 0.f;

  for (int c = 0; c < 2; ++c) {
    const int wb = u - 1 + c;
    if (wb >= 0) {
      const int tk = wb * 128 + i;
      const size_t ko = ((size_t)(b * Tn + tk)) * 512 + h * 64;
      #pragma unroll
      for (int e = 0; e < 64; ++e) {
        kch[i][e] = f2bf(kb[ko + e]);
        vch[i][e] = f2bf(vb[ko + e]);
      }
      kmask[i] = keep[b * Tn + tk] ? 3 : 1;
    } else {
      const unsigned short negone = f2bf(-1.0f);
      #pragma unroll
      for (int e = 0; e < 64; ++e) { kch[i][e] = 0; vch[i][e] = negone; }
      kmask[i] = 0;
    }
    __syncthreads();

    for (int j = 0; j < 128; ++j) {
      const int tk = wb * 128 + j;
      float d = 0.f;
      bool allowed;
      if (keepq) {
        allowed = (kmask[j] == 3) && (tk <= t);
        if (allowed) {
          float dot = 0.f;
          #pragma unroll
          for (int e = 0; e < 64; ++e) dot += q[e] * bf2f(kch[j][e]);
          d = dot * 0.125f;   // 64^-0.5
        }
      } else {
        allowed = true;       // masked query -> uniform softmax (matches ref)
      }
      if (allowed) {
        if (d > m) {
          const float sc = __expf(m - d);
          s *= sc;
          #pragma unroll
          for (int e = 0; e < 64; ++e) acc[e] *= sc;
          m = d;
        }
        const float p = __expf(d - m);
        s += p;
        #pragma unroll
        for (int e = 0; e < 64; ++e) acc[e] += p * bf2f(vch[j][e]);
      }
    }
    __syncthreads();
  }
  const float inv = (s > 0.f) ? 1.f / s : 0.f;
  #pragma unroll
  for (int e = 0; e < 64; ++e) attn[qoff + e] = f2bf(acc[e] * inv);
}

// ---------------------------------------------------------------------------
// Block-causal linear attention, heads 4..7; writes bf16 output.
// ---------------------------------------------------------------------------
__global__ __launch_bounds__(256)
void linear_attn(const float* __restrict__ qb, const float* __restrict__ kb,
                 const float* __restrict__ vb, const int* __restrict__ keep,
                 unsigned short* __restrict__ attn) {
  __shared__ float S[64 * 64];
  __shared__ float zz[64];
  __shared__ float qs[8 * 64];
  __shared__ float kbl[8 * 64];
  __shared__ float vbl[8 * 64];
  __shared__ float den[8];

  const int h = 4 + blockIdx.x;
  const int b = blockIdx.y;
  const int tid = threadIdx.x;

  for (int i = tid; i < 64 * 64; i += 256) S[i] = 0.f;
  if (tid < 64) zz[tid] = 0.f;
  __syncthreads();

  for (int u = 0; u < 128; ++u) {
    for (int i = tid; i < 512; i += 256) {
      const int p = i >> 6, e = i & 63;
      const int t = u * 8 + p;
      const size_t off = ((size_t)(b * Tn + t)) * 512 + h * 64 + e;
      const int kp = keep[b * Tn + t];
      qs[i]  = qb[off];
      kbl[i] = kp ? __expf(kb[off]) : 0.f;
      vbl[i] = kp ? vb[off] : 0.f;
    }
    __syncthreads();
    if (tid < 8) {
      const int p = tid;
      float mx = -3.0e38f;
      for (int e = 0; e < 64; ++e) mx = fmaxf(mx, qs[p * 64 + e]);
      float sum = 0.f;
      for (int e = 0; e < 64; ++e) {
        const float x = __expf(qs[p * 64 + e] - mx);
        qs[p * 64 + e] = x;
        sum += x;
      }
      const float sc = 0.125f / sum;
      float dn = 0.f;
      for (int e = 0; e < 64; ++e) {
        qs[p * 64 + e] *= sc;
        dn += qs[p * 64 + e] * zz[e];
      }
      den[p] = 1.0f / fmaxf(dn, 0.001f);
    }
    __syncthreads();
    for (int i = tid; i < 512; i += 256) {
      const int p = i >> 6, e = i & 63;
      float o = 0.f;
      for (int d = 0; d < 64; ++d) o += qs[p * 64 + d] * S[d * 64 + e];
      const int t = u * 8 + p;
      attn[((size_t)(b * Tn + t)) * 512 + h * 64 + e] = f2bf(o * den[p]);
    }
    __syncthreads();
    for (int i = tid; i < 4096; i += 256) {
      const int d = i >> 6, e = i & 63;
      float sacc = S[i];
      #pragma unroll
      for (int p = 0; p < 8; ++p) sacc += kbl[p * 64 + d] * vbl[p * 64 + e];
      S[i] = sacc;
    }
    if (tid < 64) {
      float za = zz[tid];
      #pragma unroll
      for (int p = 0; p < 8; ++p) za += kbl[p * 64 + tid];
      zz[tid] = za;
    }
    __syncthreads();
  }
}

// ---------------------------------------------------------------------------
// x = LayerNorm(x + r); writes fp32 master and bf16 copy
// ---------------------------------------------------------------------------
__global__ __launch_bounds__(256)
void residual_ln(float* __restrict__ x, const float* __restrict__ rsd,
                 const float* __restrict__ g, const float* __restrict__ bb,
                 unsigned short* __restrict__ xb) {
  __shared__ float red[256];
  const int row = blockIdx.x, tid = threadIdx.x;
  const size_t base = (size_t)row * 512;
  const float v0 = x[base + tid] + rsd[base + tid];
  const float v1 = x[base + tid + 256] + rsd[base + tid + 256];
  red[tid] = v0 + v1;
  __syncthreads();
  for (int off = 128; off > 0; off >>= 1) {
    if (tid < off) red[tid] += red[tid + off];
    __syncthreads();
  }
  const float mean = red[0] * (1.0f / 512.0f);
  __syncthreads();
  const float d0 = v0 - mean, d1 = v1 - mean;
  red[tid] = d0 * d0 + d1 * d1;
  __syncthreads();
  for (int off = 128; off > 0; off >>= 1) {
    if (tid < off) red[tid] += red[tid + off];
    __syncthreads();
  }
  const float inv = rsqrtf(red[0] * (1.0f / 512.0f) + 1e-5f);
  const float y0 = d0 * inv * g[tid] + bb[tid];
  const float y1 = d1 * inv * g[tid + 256] + bb[tid + 256];
  x[base + tid]        = y0;
  x[base + tid + 256]  = y1;
  xb[base + tid]       = f2bf(y0);
  xb[base + tid + 256] = f2bf(y1);
}

} // anonymous namespace

// ---------------------------------------------------------------------------
extern "C" void kernel_launch(void* const* d_in, const int* in_sizes, int n_in,
                              void* d_out, int out_size, void* d_ws, size_t ws_size,
                              hipStream_t stream) {
  (void)in_sizes; (void)n_in; (void)out_size; (void)ws_size;

  const int*   x_mcc  = (const int*)d_in[0];
  const int*   x_ab   = (const int*)d_in[1];
  const int*   x_hour = (const int*)d_in[2];
  const float* e_mcc  = (const float*)d_in[3];
  const float* e_ab   = (const float*)d_in[4];
  const float* e_hr   = (const float*)d_in[5];
  const float* proj_w = (const float*)d_in[6];
  const float* proj_b = (const float*)d_in[7];
  const float* Wq     = (const float*)d_in[8];
  const float* Wk     = (const float*)d_in[9];
  const float* Wv     = (const float*)d_in[10];
  const float* Wo     = (const float*)d_in[11];
  const float* bo     = (const float*)d_in[12];
  const float* ff1_w  = (const float*)d_in[13];
  const float* ff1_b  = (const float*)d_in[14];
  const float* ff2_w  = (const float*)d_in[15];
  const float* ff2_b  = (const float*)d_in[16];
  const float* ln1_g  = (const float*)d_in[17];
  const float* ln1_b  = (const float*)d_in[18];
  const float* ln2_g  = (const float*)d_in[19];
  const float* ln2_b  = (const float*)d_in[20];

  // ---- workspace carving (64B aligned) ------------------------------------
  char* wsb = (char*)d_ws;
  size_t off = 0;
  auto alloc = [&](size_t bytes) -> void* {
    void* p = wsb + off;
    off += bytes;
    off = (off + 63) & ~(size_t)63;
    return p;
  };
  float* xbuf  = (float*)alloc((size_t)Mrows * Dn * 4);
  float* tmp   = (float*)alloc((size_t)Mrows * Dn * 4);
  float* qbuf  = (float*)alloc((size_t)Mrows * Dn * 4);
  float* kbuf  = (float*)alloc((size_t)Mrows * Dn * 4);
  float* vbuf  = (float*)alloc((size_t)Mrows * Dn * 4);
  unsigned short* xb    = (unsigned short*)alloc((size_t)Mrows * Dn * 2);
  unsigned short* xin   = (unsigned short*)alloc((size_t)Mrows * 224 * 2);
  unsigned short* attnb = (unsigned short*)alloc((size_t)Mrows * Dn * 2);
  unsigned short* ffhb  = (unsigned short*)alloc((size_t)Mrows * FFn * 2);
  unsigned short* hhb   = (unsigned short*)alloc((size_t)Mrows * Dn * 2);
  int* keepm = (int*)alloc((size_t)Mrows * 4);

  const dim3 blk256(256);
  const dim3 gN512(4, 128);    // N=512
  const dim3 gN2048(16, 128);  // N=2048

  // weight prep helper: fp32 [K x N] -> bf16 [Npad x K]
  auto prep = [&](const float* W, int K, int N) -> const unsigned short* {
    const int Npad = ((N + 127) / 128) * 128;
    unsigned short* t = (unsigned short*)alloc((size_t)Npad * K * 2);
    const int total = Npad * K;
    convert_transpose<<<(total + 255) / 256, blk256, 0, stream>>>(W, t, K, N, Npad);
    return t;
  };

  // ---- embed + input projection + positional encoding ----------------------
  embed_kernel<<<Mrows, 224, 0, stream>>>(x_mcc, x_ab, x_hour,
                                          e_mcc, e_ab, e_hr, xin, keepm);
  const unsigned short* projT = prep(proj_w, 224, 512);
  gemm_bf16_wmma<true, false, false><<<gN512, blk256, 0, stream>>>(
      xin, projT, proj_b, xbuf, nullptr, 512, 224);
  add_pos<<<(Mrows * Dn) / 256, blk256, 0, stream>>>(xbuf, xb);

  // ---- transformer layers --------------------------------------------------
  for (int l = 0; l < Ln; ++l) {
    const unsigned short* wqT  = prep(Wq + (size_t)l * Dn * Dn, 512, 512);
    const unsigned short* wkT  = prep(Wk + (size_t)l * Dn * Dn, 512, 512);
    const unsigned short* wvT  = prep(Wv + (size_t)l * Dn * Dn, 512, 512);
    const unsigned short* woT  = prep(Wo + (size_t)l * Dn * Dn, 512, 512);
    const unsigned short* ff1T = prep(ff1_w + (size_t)l * Dn * FFn, 512, 2048);
    const unsigned short* ff2T = prep(ff2_w + (size_t)l * FFn * Dn, 2048, 512);

    gemm_bf16_wmma<false, false, false><<<gN512, blk256, 0, stream>>>(
        xb, wqT, nullptr, qbuf, nullptr, 512, 512);
    gemm_bf16_wmma<false, false, false><<<gN512, blk256, 0, stream>>>(
        xb, wkT, nullptr, kbuf, nullptr, 512, 512);
    gemm_bf16_wmma<false, false, false><<<gN512, blk256, 0, stream>>>(
        xb, wvT, nullptr, vbuf, nullptr, 512, 512);

    local_attn<<<dim3(8, 4, 16), 128, 0, stream>>>(qbuf, kbuf, vbuf, keepm, attnb);
    linear_attn<<<dim3(4, 16), blk256, 0, stream>>>(qbuf, kbuf, vbuf, keepm, attnb);

    gemm_bf16_wmma<true, false, false><<<gN512, blk256, 0, stream>>>(
        attnb, woT, bo + (size_t)l * Dn, tmp, nullptr, 512, 512);
    residual_ln<<<Mrows, blk256, 0, stream>>>(xbuf, tmp,
        ln1_g + (size_t)l * Dn, ln1_b + (size_t)l * Dn, xb);

    gemm_bf16_wmma<true, true, true><<<gN2048, blk256, 0, stream>>>(
        xb, ff1T, ff1_b + (size_t)l * FFn, nullptr, ffhb, 2048, 512);
    gemm_bf16_wmma<true, false, false><<<gN512, blk256, 0, stream>>>(
        ffhb, ff2T, ff2_b + (size_t)l * Dn, tmp, nullptr, 512, 2048);
    residual_ln<<<Mrows, blk256, 0, stream>>>(xbuf, tmp,
        ln2_g + (size_t)l * Dn, ln2_b + (size_t)l * Dn, xb);
  }

  // ---- output heads --------------------------------------------------------
  float* out = (float*)d_out;
  const float* hd1w[3] = {(const float*)d_in[21], (const float*)d_in[25], (const float*)d_in[29]};
  const float* hd1b[3] = {(const float*)d_in[22], (const float*)d_in[26], (const float*)d_in[30]};
  const float* hd2w[3] = {(const float*)d_in[23], (const float*)d_in[27], (const float*)d_in[31]};
  const float* hd2b[3] = {(const float*)d_in[24], (const float*)d_in[28], (const float*)d_in[32]};
  const int vocab[3] = {400, 128, 24};

  size_t ooff = 0;
  for (int hj = 0; hj < 3; ++hj) {
    const unsigned short* h1T = prep(hd1w[hj], 512, 512);
    const unsigned short* h2T = prep(hd2w[hj], 512, vocab[hj]);
    gemm_bf16_wmma<true, true, true><<<gN512, blk256, 0, stream>>>(
        xb, h1T, hd1b[hj], nullptr, hhb, 512, 512);
    const dim3 gHead((vocab[hj] + 127) / 128, 128);
    gemm_bf16_wmma<true, false, false><<<gHead, blk256, 0, stream>>>(
        hhb, h2T, hd2b[hj], out + ooff, nullptr, vocab[hj], 512);
    ooff += (size_t)Mrows * vocab[hj];
  }
}
